// WaterNet0510_74835510166091
// MI455X (gfx1250) — matrix operand order repaired and checked by reading, not compiled, not run on hardware
//
#include <hip/hip_runtime.h>
#include <math.h>

#define NT 365
#define NS 128
#define NG 16
#define NH 8
#define HS 256
#define EPSV 1e-6f

typedef __attribute__((ext_vector_type(16))) _Float16 v16h;
typedef __attribute__((ext_vector_type(8)))  float    v8f;

// Branchless tanh: 1 - 2/(1+exp(2z)) with native exp2/rcp.
// exp2(+inf)->inf => 1 ; exp2(-inf)->0 => -1. No EXEC divergence.
__device__ __forceinline__ float fast_tanh(float z) {
    float e = __builtin_amdgcn_exp2f(z * 2.8853900817779268f);  // 2*log2(e)
    float r = __builtin_amdgcn_rcpf(e + 1.0f);
    return 1.0f - 2.0f * r;
}

// ---------------- Phase A: state = xc @ W_fc^T + b_fc ; state_b = state + b_kin --------
__global__ void state_kernel(const float* __restrict__ xc, const float* __restrict__ Wfc,
                             const float* __restrict__ bfc, const float* __restrict__ bkin,
                             float* __restrict__ state, float* __restrict__ stateb) {
    int idx = blockIdx.x * blockDim.x + threadIdx.x;
    if (idx >= NS * HS) return;
    int s = idx >> 8;          // / HS
    int k = idx & (HS - 1);
    float acc = bfc[k];
    const float* xr = xc + s * NG;
    const float* wr = Wfc + k * NG;
#pragma unroll
    for (int g = 0; g < NG; ++g) acc += xr[g] * wr[g];
    state[idx]  = acc;
    stateb[idx] = acc + bkin[k];
}

// ---------------- Phase A: pg = tanh(state) @ W_g^T + b_g, then gates ----------------
__global__ void gates_kernel(const float* __restrict__ state, const float* __restrict__ Wg,
                             const float* __restrict__ bg,
                             float* __restrict__ gk, float* __restrict__ gl,
                             float* __restrict__ qb, float* __restrict__ ga,
                             float* __restrict__ gi, float* __restrict__ ge) {
    __shared__ float ts[HS];
    __shared__ float pg[6 * NH];
    int s = blockIdx.x;
    int tid = threadIdx.x;
    ts[tid] = tanhf(state[s * HS + tid]);
    __syncthreads();
    if (tid < 6 * NH) {
        float acc = bg[tid];
        const float* wr = Wg + tid * HS;
        for (int k = 0; k < HS; ++k) acc += wr[k] * ts[k];
        pg[tid] = acc;
    }
    __syncthreads();
    if (tid < NH) {
        int h = tid;
        gk[s * NH + h] = expf(pg[0 * NH + h]) * 0.01f;
        gl[s * NH + h] = expf(pg[1 * NH + h]) * 100.0f;
        qb[s * NH + h] = fmaxf(pg[2 * NH + h], 0.0f) * 0.1f;
        float mx = -1e30f;
        for (int j = 0; j < NH; ++j) mx = fmaxf(mx, pg[3 * NH + j]);
        float sum = 0.0f;
        for (int j = 0; j < NH; ++j) sum += expf(pg[3 * NH + j] - mx);
        ga[s * NH + h] = expf(pg[3 * NH + h] - mx) / sum;
        float hsg = fminf(fmaxf(pg[4 * NH + h] * (1.0f / 6.0f) + 0.5f, 0.0f), 1.0f);
        gi[s * NH + h] = hsg * 0.5f;
        ge[s * NH + h] = fmaxf(pg[5 * NH + h], 0.0f);
    }
}

// ---------------- _divide_p precompute ----------------
__global__ void psl_kernel(const float* __restrict__ x, float* __restrict__ ps,
                           float* __restrict__ pl) {
    int i = blockIdx.x * blockDim.x + threadIdx.x;
    if (i >= NT * NS) return;
    const float* xr = x + i * 6;
    float p = xr[0], t1 = xr[2], t2 = xr[3];
    float r = (t1 + t2) / fmaxf(t2 - t1, EPSV);
    r = fminf(fmaxf(r, -1.0f), 1.0f);
    float vf = acosf(r) * 0.31830988618379067f;  // 1/pi
    if (t1 >= 0.0f) vf = 0.0f;
    if (t2 <= 0.0f) vf = 1.0f;
    ps[i] = p * vf;
    pl[i] = p * (1.0f - vf);
}

// ---------------- Phase B: km = exp(tanh(F@Wkin^T + state_b) @ Wkout^T + bkout) --------
// One wave per 16-row M-tile; K=256 fully unrolled -> 8x v_wmma_f32_16x16x32_f16.
// LDS: Wkin rows (float4, 4KB) + 8 pre-converted f16 B fragments (8KB), shared by 8 waves.
__global__ void km_kernel(const float* __restrict__ x, const float* __restrict__ stateb,
                          const float* __restrict__ Wkin,
                          const float* __restrict__ Wkout, const float* __restrict__ bkout,
                          float* __restrict__ km) {
    __shared__ float4 sWk[HS];     // Wkin[k][0..3]
    __shared__ v16h  sB[8][32];    // B fragment per k-step per lane
    int tid = threadIdx.x;

    // Cooperative staging (256 threads: one Wkin row + one B-fragment entry each).
    sWk[tid] = *(const float4*)(Wkin + tid * 4);
    {
        int step = tid >> 5, lane = tid & 31;
        int n = lane & 15, hi = lane >> 4;
        int k0 = step * 32;
#pragma unroll
        for (int e = 0; e < 16; ++e) {
            int kb = k0 + e + 8 * hi + ((e >= 8) ? 8 : 0);   // B layout (16-bit, 32x16): K = e + 8*hi + 8*(e>=8), N = lane&15
            sB[step][lane][e] = (n < NH) ? (_Float16)Wkout[n * HS + kb] : (_Float16)0.0f;
        }
    }
    __syncthreads();

    int wave = blockIdx.x * (blockDim.x >> 5) + (tid >> 5);
    int lane = tid & 31;
    int row0 = wave * 16;
    int m  = lane & 15;
    int hi = lane >> 4;
    int row = row0 + m;                 // row = t*NS + s
    int s = row & (NS - 1);
    const float* xr = x + row * 6;
    float f0 = xr[2], f1 = xr[3], f2 = xr[4], f3 = xr[5];
    const float* stb = stateb + s * HS;
    int n = lane & 15;

    v8f c = {};
#pragma unroll
    for (int step = 0; step < 8; ++step) {
        int k0 = step * 32;
        v16h a;
#pragma unroll
        for (int e = 0; e < 16; ++e) {
            // A layout (16-bit, 16x32): lane m = l&15; K = e + 8*hi + 8*(e>=8)
            int ka = k0 + e + 8 * hi + ((e >= 8) ? 8 : 0);
            float4 wv = sWk[ka];
            float z = f0 * wv.x + f1 * wv.y + f2 * wv.z + f3 * wv.w + stb[ka];
            a[e] = (_Float16)fast_tanh(z);
        }
        v16h b = sB[step][lane];
        c = __builtin_amdgcn_wmma_f32_16x16x32_f16(false, a, false, b,
                                                   (short)0, c, false, false);
    }
    if (n < NH) {
        float bk = bkout[n];
#pragma unroll
        for (int r = 0; r < 8; ++r) {
            int mm = r + 8 * hi;        // C/D layout: VGPR r -> M = r + 8*hi, N = lane&15
            km[(row0 + mm) * NH + n] = expf(c[r] + bk);
        }
    }
}

// ---------------- Phase C: collapsed scan, one lane per (site,bucket) ----------------
// Delay line reduces exactly to its running sum S (uniform depth scalings, mass-
// conserving shift, sum-only outputs). q reduced over NH=8 lanes via shfl_xor.
__global__ void scan_kernel(const float* __restrict__ x, const float* __restrict__ ps,
                            const float* __restrict__ pl, const float* __restrict__ km,
                            const float* __restrict__ gk, const float* __restrict__ gl,
                            const float* __restrict__ qb, const float* __restrict__ ga,
                            const float* __restrict__ gi, const float* __restrict__ ge,
                            float* __restrict__ q_out) {
    int g = blockIdx.x * blockDim.x + threadIdx.x;
    if (g >= NS * NH) return;
    int s = g >> 3;
    int h = g & (NH - 1);
    float sf = 0.0f, S = 0.0f;
    float gkv = gk[g], glv = gl[g], qbv = qb[g];
    float gav = ga[g], giv = gi[g], gev = ge[g];
    for (int t = 0; t < NT; ++t) {
        int i = t * NS + s;
        float ps_t = ps[i];
        float pl_t = pl[i];
        float ev_t = x[i * 6 + 1];
        float kmv  = km[i * NH + h];

        float sf1  = sf + ps_t;
        float melt = fminf(sf1, kmv);
        sf = sf1 - melt;
        float w_in  = melt + pl_t * giv;
        float e_pot = ev_t * gev;
        float S_   = S + w_in;                                  // s = w_in + sum(d)
        float f_ev = fminf(fmaxf(e_pot / fmaxf(S_, EPSV), 0.0f), 1.0f);
        float s1   = S_ * (1.0f - f_ev);
        float q_s  = fmaxf(s1 - glv, 0.0f);
        float s2   = s1 - q_s;
        float q_sub = fminf(s2 * gkv + qbv, s2);
        float f_q  = q_sub / fmaxf(s2, EPSV);
        float qf   = f_q * s2;                                  // sum(d2*f_q)
        S = s2 - qf;                                            // sum(d_new)

        float qv = (qf + q_s) * gav;
        qv += __shfl_xor(qv, 1, 32);
        qv += __shfl_xor(qv, 2, 32);
        qv += __shfl_xor(qv, 4, 32);
        if (h == 0) q_out[i] = qv;
    }
}

extern "C" void kernel_launch(void* const* d_in, const int* in_sizes, int n_in,
                              void* d_out, int out_size, void* d_ws, size_t ws_size,
                              hipStream_t stream) {
    const float* x     = (const float*)d_in[0];
    const float* xc    = (const float*)d_in[1];
    const float* Wfc   = (const float*)d_in[2];
    const float* bfc   = (const float*)d_in[3];
    const float* Wg    = (const float*)d_in[4];
    const float* bg    = (const float*)d_in[5];
    const float* Wkin  = (const float*)d_in[6];
    const float* bkin  = (const float*)d_in[7];
    const float* Wkout = (const float*)d_in[8];
    const float* bkout = (const float*)d_in[9];
    float* out = (float*)d_out;

    float* wsf    = (float*)d_ws;
    float* state  = wsf;  wsf += NS * HS;
    float* stateb = wsf;  wsf += NS * HS;
    float* gk     = wsf;  wsf += NS * NH;
    float* gl     = wsf;  wsf += NS * NH;
    float* qb     = wsf;  wsf += NS * NH;
    float* ga     = wsf;  wsf += NS * NH;
    float* gi     = wsf;  wsf += NS * NH;
    float* ge     = wsf;  wsf += NS * NH;
    float* km     = wsf;  wsf += NT * NS * NH;
    float* ps     = wsf;  wsf += NT * NS;
    float* pl     = wsf;  wsf += NT * NS;

    state_kernel<<<(NS * HS + 255) / 256, 256, 0, stream>>>(xc, Wfc, bfc, bkin,
                                                            state, stateb);
    gates_kernel<<<NS, HS, 0, stream>>>(state, Wg, bg, gk, gl, qb, ga, gi, ge);
    psl_kernel<<<(NT * NS + 255) / 256, 256, 0, stream>>>(x, ps, pl);
    // 2920 M-tiles of 16 rows: 365 blocks x 8 waves (wave32), EXEC fully active.
    km_kernel<<<NT, 256, 0, stream>>>(x, stateb, Wkin, Wkout, bkout, km);
    scan_kernel<<<(NS * NH + 255) / 256, 256, 0, stream>>>(x, ps, pl, km,
                                                           gk, gl, qb, ga, gi, ge, out);
}
